// Model_20770461843918
// MI455X (gfx1250) — matrix operand order repaired and checked by reading, not compiled, not run on hardware
//
#include <hip/hip_runtime.h>

// Problem constants (from the reference)
#define BATCH 4
#define LEN   1024
#define CH    128      // C
#define DI    256      // 2*C
#define NS    16       // N (state dim)
#define RR    8        // R (dt rank)
#define KC    4        // depthwise conv kernel
#define XDB_LD 48      // padded row stride for xdb (R + 2N = 40 -> 48)

typedef __attribute__((ext_vector_type(16))) _Float16 v16h;
typedef __attribute__((ext_vector_type(8)))  _Float16 v8h;
typedef __attribute__((ext_vector_type(4)))  _Float16 v4h;
typedef __attribute__((ext_vector_type(8)))  float    v8f;

enum { ACT_NONE = 0, ACT_SIGMOID = 2, ACT_SOFTPLUS = 3 };

__device__ __forceinline__ float dev_sigmoid(float x) { return 1.0f / (1.0f + __expf(-x)); }
__device__ __forceinline__ float dev_silu(float x)    { return x * dev_sigmoid(x); }
__device__ __forceinline__ float dev_softplus(float x){ return (x > 20.0f) ? x : log1pf(__expf(x)); }

__device__ __forceinline__ v16h cat8(v8h lo, v8h hi)
{
    return __builtin_shufflevector(lo, hi, 0,1,2,3,4,5,6,7,8,9,10,11,12,13,14,15);
}

// ---------------------------------------------------------------------------
// Guard-free f16 GEMM:  Y[M,Nn](f32) = act( X[M,K](f16,ldx) @ W[Nn,K](f16)^T + bias )
// Preconditions: M % 64 == 0, Nn % 16 == 0, Kdim % 32 == 0, strides 16B-aligned.
//
// One wave per 64x16 strip (4 accumulators share one B fragment).
// CDNA5 ISA 7.12.2 wave32 striping:
//  A 16x32 f16: lane half hh, elem i<8 -> K = 8*hh+i ; i>=8 -> K = 16+8*hh+(i-8)
//    => two contiguous 8-half runs -> two 16B v8h loads
//  B 32x16 f16: same K striping as A => two contiguous 8-half runs -> two 16B v8h loads
//  C/D 16x16 f32: VGPR r -> row r + 8*hh, col = lane&15
// All 10 fragment loads per K-step are issued before the 4 WMMAs (one wait).
// ---------------------------------------------------------------------------
template<int ACT>
__global__ __launch_bounds__(32)
void gemm_xwt_wmma(const _Float16* __restrict__ X, int ldx,
                   const _Float16* __restrict__ W,      // [Nn, Kdim] row-major
                   const float* __restrict__ bias,      // [Nn] or nullptr
                   float* __restrict__ Y, int ldy,
                   int Kdim)
{
    const int lane = threadIdx.x & 31;
    const int hh   = lane >> 4;
    const int lid  = lane & 15;
    const int n0   = blockIdx.x << 4;
    const int m0   = blockIdx.y << 6;

    const _Float16* wrow = W + (size_t)(n0 + lid) * Kdim;
    const _Float16* xr0  = X + (size_t)(m0 + lid) * ldx;
    const size_t tstep = (size_t)16 * ldx;

    v8f acc[4] = {};
    for (int kk = 0; kk < Kdim; kk += 32) {
        if (kk + 32 < Kdim) {                     // gfx1250 global_prefetch_b8
            __builtin_prefetch(wrow + kk + 32, 0, 0);
            __builtin_prefetch(xr0 + kk + 32, 0, 0);
        }
        const _Float16* bp  = wrow + kk + 8 * hh;
        const _Float16* ap  = xr0 + kk + 8 * hh;
        const v16h b  = cat8(*(const v8h*)bp,                  *(const v8h*)(bp + 16));
        const v16h a0 = cat8(*(const v8h*)(ap),                *(const v8h*)(ap + 16));
        const v16h a1 = cat8(*(const v8h*)(ap + tstep),        *(const v8h*)(ap + tstep + 16));
        const v16h a2 = cat8(*(const v8h*)(ap + 2 * tstep),    *(const v8h*)(ap + 2 * tstep + 16));
        const v16h a3 = cat8(*(const v8h*)(ap + 3 * tstep),    *(const v8h*)(ap + 3 * tstep + 16));
        acc[0] = __builtin_amdgcn_wmma_f32_16x16x32_f16(false, a0, false, b, (short)0, acc[0], false, false);
        acc[1] = __builtin_amdgcn_wmma_f32_16x16x32_f16(false, a1, false, b, (short)0, acc[1], false, false);
        acc[2] = __builtin_amdgcn_wmma_f32_16x16x32_f16(false, a2, false, b, (short)0, acc[2], false, false);
        acc[3] = __builtin_amdgcn_wmma_f32_16x16x32_f16(false, a3, false, b, (short)0, acc[3], false, false);
    }

    const int col = n0 + lid;
    const float bv = bias ? bias[col] : 0.0f;
#pragma unroll
    for (int t = 0; t < 4; ++t) {
#pragma unroll
        for (int r = 0; r < 8; ++r) {
            const int row = m0 + 16 * t + r + 8 * hh;
            float vv = acc[t][r] + bv;
            if (ACT == ACT_SIGMOID)       vv = dev_sigmoid(vv);
            else if (ACT == ACT_SOFTPLUS) vv = dev_softplus(vv);
            Y[(size_t)row * ldy + col] = vv;
        }
    }
}

// ---------------------------------------------------------------------------
// f32 -> f16 bulk convert, 4 elements/thread (total % 4 == 0).
// ---------------------------------------------------------------------------
__global__ void cvt_f32_f16_kernel(const float* __restrict__ src, _Float16* __restrict__ dst, int total4)
{
    const int idx = blockIdx.x * blockDim.x + threadIdx.x;
    if (idx >= total4) return;
    const float4 v = *(const float4*)(src + (size_t)idx * 4);
    v4h o; o[0] = (_Float16)v.x; o[1] = (_Float16)v.y; o[2] = (_Float16)v.z; o[3] = (_Float16)v.w;
    *(v4h*)(dst + (size_t)idx * 4) = o;
}

// Pad+convert: Wx (7,40,256) -> f16 (7,48,256) zero rows
__global__ void pad_wx_kernel(const float* __restrict__ wx, _Float16* __restrict__ out, int total)
{
    const int idx = blockIdx.x * blockDim.x + threadIdx.x;
    if (idx >= total) return;
    const int k = idx & 255;
    const int r = (idx >> 8) % XDB_LD;
    const int b = idx / (XDB_LD * 256);
    out[idx] = (r < RR + 2 * NS) ? (_Float16)wx[((size_t)b * (RR + 2 * NS) + r) * 256 + k] : (_Float16)0.0f;
}

// Pad+convert: Wdt (7,256,8) -> f16 (7,256,32) zero cols
__global__ void pad_wdt_kernel(const float* __restrict__ wdt, _Float16* __restrict__ out, int total)
{
    const int idx = blockIdx.x * blockDim.x + threadIdx.x;
    if (idx >= total) return;
    const int k = idx & 31;
    const int d = (idx >> 5) & (DI - 1);
    const int b = idx / (DI * 32);
    out[idx] = (k < RR) ? (_Float16)wdt[((size_t)b * DI + d) * RR + k] : (_Float16)0.0f;
}

// dtin_h[m, 0:32] = f16([xdb[m, 0:8], zeros]) so the dt GEMM runs with K=32
__global__ void dtin_kernel(const float* __restrict__ xdb, _Float16* __restrict__ dtin, int total)
{
    const int idx = blockIdx.x * blockDim.x + threadIdx.x;
    if (idx >= total) return;
    const int k = idx & 31;
    const int m = idx >> 5;
    dtin[idx] = (k < RR) ? (_Float16)xdb[(size_t)m * XDB_LD + k] : (_Float16)0.0f;
}

// ---------------------------------------------------------------------------
// Depthwise causal conv (K=4, left pad 3) + SiLU.
// Writes u in f32 (for the scan) and f16 (for the Wx GEMM).
// ---------------------------------------------------------------------------
__global__ void dwconv_silu_kernel(const float* __restrict__ xz,
                                   const float* __restrict__ cw,
                                   const float* __restrict__ cb,
                                   float* __restrict__ u, _Float16* __restrict__ uh,
                                   int Lc, int total)
{
    const int idx = blockIdx.x * blockDim.x + threadIdx.x;
    if (idx >= total) return;
    const int d = idx & (DI - 1);
    const int m = idx / DI;
    const int l = m % Lc;
    float s = cb[d];
#pragma unroll
    for (int k = 0; k < KC; ++k) {
        const int lp = l + k - (KC - 1);
        if (lp >= 0) s += cw[d * KC + k] * xz[(size_t)(m + lp - l) * (2 * DI) + d];
    }
    const float r = dev_silu(s);
    u[(size_t)m * DI + d]  = r;
    uh[(size_t)m * DI + d] = (_Float16)r;
}

// ---------------------------------------------------------------------------
// Selective scan. One block per batch, DI threads; Bm/Cm via LDS each step.
// ---------------------------------------------------------------------------
__global__ __launch_bounds__(DI)
void scan_kernel(const float* __restrict__ u,     // [B*Lc, DI]
                 const float* __restrict__ dt,    // [B*Lc, DI]
                 const float* __restrict__ xdb,   // [B*Lc, XDB_LD]
                 const float* __restrict__ Alog,  // [DI, NS]
                 const float* __restrict__ Dp,    // [DI]
                 float* __restrict__ y, int Lc)
{
    const int b = blockIdx.x;
    const int d = threadIdx.x;
    __shared__ float sB[NS];
    __shared__ float sC[NS];
    float h[NS], A[NS];
#pragma unroll
    for (int n = 0; n < NS; ++n) { h[n] = 0.0f; A[n] = -__expf(Alog[d * NS + n]); }
    const float Dv = Dp[d];

    for (int l = 0; l < Lc; ++l) {
        const size_t m = (size_t)b * Lc + l;
        if (d < 2 * NS) {
            const float v = xdb[m * XDB_LD + RR + d];
            if (d < NS) sB[d] = v; else sC[d - NS] = v;
        }
        __syncthreads();
        const float dtv = dt[m * DI + d];
        const float uv  = u[m * DI + d];
        const float du  = dtv * uv;
        float acc = 0.0f;
#pragma unroll
        for (int n = 0; n < NS; ++n) {
            const float dA = __expf(dtv * A[n]);
            h[n] = dA * h[n] + du * sB[n];
            acc += h[n] * sC[n];
        }
        y[m * DI + d] = acc + uv * Dv;
        __syncthreads();
    }
}

// yz = f16( y * silu(z) ), z = xz[:, DI:2*DI]
__global__ void ygate_kernel(const float* __restrict__ y, const float* __restrict__ xz,
                             _Float16* __restrict__ yz, int total)
{
    const int idx = blockIdx.x * blockDim.x + threadIdx.x;
    if (idx >= total) return;
    const int d = idx & (DI - 1);
    const int m = idx / DI;
    yz[idx] = (_Float16)(y[idx] * dev_silu(xz[(size_t)m * (2 * DI) + DI + d]));
}

// Strided downconv: out[b,lo,co] = b[co] + sum_{k,ci} w[co,ci,k]*in[b,2lo+k-1,ci]
__global__ void downconv_kernel(const float* __restrict__ in, const float* __restrict__ w,
                                const float* __restrict__ bsc, float* __restrict__ out,
                                int Li, int Lo, int total)
{
    const int idx = blockIdx.x * blockDim.x + threadIdx.x;
    if (idx >= total) return;
    const int co = idx % CH;
    const int lo = (idx / CH) % Lo;
    const int b  = idx / (CH * Lo);
    float s = bsc[co];
#pragma unroll
    for (int k = 0; k < 3; ++k) {
        const int pos = 2 * lo + k - 1;
        if (pos < 0 || pos >= Li) continue;
        const float* row = in + (size_t)(b * Li + pos) * CH;
        const float* wr  = w + (size_t)co * CH * 3 + k;
        float acc = 0.0f;
        for (int ci = 0; ci < CH; ++ci) acc += wr[ci * 3] * row[ci];
        s += acc;
    }
    out[(size_t)(b * Lo + lo) * CH + co] = s;
}

// Transposed-conv x2 upsample (kernel 2, stride 2)
__global__ void upsample_kernel(const float* __restrict__ t2, const float* __restrict__ uw,
                                const float* __restrict__ ub, float* __restrict__ t2u,
                                int Li, int Lo, int total)
{
    const int idx = blockIdx.x * blockDim.x + threadIdx.x;
    if (idx >= total) return;
    const int o = idx % CH;
    const int l = (idx / CH) % Lo;
    const int b = idx / (CH * Lo);
    const int par  = l & 1;
    const int mpos = l >> 1;
    const float* row = t2 + (size_t)(b * Li + mpos) * CH;
    float s = ub[o];
    for (int i = 0; i < CH; ++i) s += uw[(i * CH + o) * 2 + par] * row[i];
    t2u[(size_t)(b * Lo + l) * CH + o] = s;
}

// fused = f16( concat([t1, t2u], channel) )
__global__ void fuse_kernel(const float* __restrict__ t1, const float* __restrict__ t2u,
                            _Float16* __restrict__ fused, int total)
{
    const int idx = blockIdx.x * blockDim.x + threadIdx.x;
    if (idx >= total) return;
    const int c = idx % (2 * CH);
    const int m = idx / (2 * CH);
    fused[idx] = (_Float16)((c < CH) ? t1[(size_t)m * CH + c] : t2u[(size_t)m * CH + c - CH]);
}

// mix = f16( concat([t1*w, t2u*(1-w)], channel) )
__global__ void mix_kernel(const float* __restrict__ t1, const float* __restrict__ t2u,
                           const float* __restrict__ w, _Float16* __restrict__ mix, int total)
{
    const int idx = blockIdx.x * blockDim.x + threadIdx.x;
    if (idx >= total) return;
    const int c = idx % (2 * CH);
    const int m = idx / (2 * CH);
    float r;
    if (c < CH) r = t1[(size_t)m * CH + c] * w[(size_t)m * CH + c];
    else        r = t2u[(size_t)m * CH + c - CH] * (1.0f - w[(size_t)m * CH + c - CH]);
    mix[idx] = (_Float16)r;
}

// ---------------------------------------------------------------------------
// Host-side orchestration
// ---------------------------------------------------------------------------
struct Ctx {
    const float *m_convw, *m_convb, *m_bdt, *m_Alog, *m_D;
    const float *dc_w, *dc_b, *wg_b, *db_b, *up_w, *up_b;
    const _Float16 *win_h, *wxp_h, *wdtp_h, *wout_h, *wg_h, *db_h;   // f16 weights (in ws)
    float *t_xz, *t_u, *t_xdb, *t_dt, *t_y;
    _Float16 *t_inh, *t_uh, *t_dtinh, *t_yzh, *g_fush, *g_mixh;
    float *g_t2u, *g_w, *g_f;
    hipStream_t stream;
};

static inline void launch_gemm(int act, const _Float16* X, int ldx, const _Float16* W,
                               const float* bias, float* Y, int ldy,
                               int M, int Nn, int Kdim, hipStream_t s)
{
    dim3 grid(Nn / 16, M / 64), blk(32);
    switch (act) {
    case ACT_SIGMOID:
        gemm_xwt_wmma<ACT_SIGMOID><<<grid, blk, 0, s>>>(X, ldx, W, bias, Y, ldy, Kdim); break;
    case ACT_SOFTPLUS:
        gemm_xwt_wmma<ACT_SOFTPLUS><<<grid, blk, 0, s>>>(X, ldx, W, bias, Y, ldy, Kdim); break;
    default:
        gemm_xwt_wmma<ACT_NONE><<<grid, blk, 0, s>>>(X, ldx, W, bias, Y, ldy, Kdim); break;
    }
}

static void run_mb(const Ctx& c, const float* in, float* out, int blk_i, int Lc)
{
    const int M = BATCH * Lc;
    const _Float16* Win  = c.win_h  + (size_t)blk_i * 2 * DI * CH;
    const _Float16* Wxp  = c.wxp_h  + (size_t)blk_i * XDB_LD * DI;
    const _Float16* Wdtp = c.wdtp_h + (size_t)blk_i * DI * 32;
    const _Float16* Wout = c.wout_h + (size_t)blk_i * CH * DI;
    const float* cw   = c.m_convw + (size_t)blk_i * DI * KC;
    const float* cb   = c.m_convb + (size_t)blk_i * DI;
    const float* bdt  = c.m_bdt   + (size_t)blk_i * DI;
    const float* Alog = c.m_Alog  + (size_t)blk_i * DI * NS;
    const float* Dp   = c.m_D     + (size_t)blk_i * DI;

    // in -> f16
    { int t4 = (M * CH) / 4;
      cvt_f32_f16_kernel<<<(t4 + 255) / 256, 256, 0, c.stream>>>(in, c.t_inh, t4); }
    // xz = x @ Win^T                                 [M, 2*DI]
    launch_gemm(ACT_NONE, c.t_inh, CH, Win, nullptr, c.t_xz, 2 * DI, M, 2 * DI, CH, c.stream);
    // u = silu(causal depthwise conv(xi) + b)        [M, DI] (f32 + f16)
    { int total = M * DI;
      dwconv_silu_kernel<<<(total + 255) / 256, 256, 0, c.stream>>>(
          c.t_xz, cw, cb, c.t_u, c.t_uh, Lc, total); }
    // xdb = u @ Wx_pad^T                             [M, 48] (cols 40..47 are 0)
    launch_gemm(ACT_NONE, c.t_uh, DI, Wxp, nullptr, c.t_xdb, XDB_LD, M, XDB_LD, DI, c.stream);
    // dtin = f16([xdb[:, :8], 0])                    [M, 32]
    { int total = M * 32;
      dtin_kernel<<<(total + 255) / 256, 256, 0, c.stream>>>(c.t_xdb, c.t_dtinh, total); }
    // dt = softplus(dtin @ Wdt_pad^T + bdt)          [M, DI]
    launch_gemm(ACT_SOFTPLUS, c.t_dtinh, 32, Wdtp, bdt, c.t_dt, DI, M, DI, 32, c.stream);
    // selective scan
    scan_kernel<<<BATCH, DI, 0, c.stream>>>(c.t_u, c.t_dt, c.t_xdb, Alog, Dp, c.t_y, Lc);
    // yz = f16(y * silu(z))
    { int total = M * DI;
      ygate_kernel<<<(total + 255) / 256, 256, 0, c.stream>>>(c.t_y, c.t_xz, c.t_yzh, total); }
    // out = yz @ Wout^T                              [M, CH]
    launch_gemm(ACT_NONE, c.t_yzh, DI, Wout, nullptr, out, CH, M, CH, DI, c.stream);
}

static void run_gate(const Ctx& c, const float* t1, int Lo, const float* t2,
                     int g, int mb_i, float* out)
{
    const int Li = Lo / 2;
    const int M  = BATCH * Lo;
    { int total = M * CH;
      upsample_kernel<<<(total + 255) / 256, 256, 0, c.stream>>>(
          t2, c.up_w + (size_t)g * CH * CH * 2, c.up_b + (size_t)g * CH, c.g_t2u, Li, Lo, total); }
    { int total = M * 2 * CH;
      fuse_kernel<<<(total + 255) / 256, 256, 0, c.stream>>>(t1, c.g_t2u, c.g_fush, total); }
    launch_gemm(ACT_SIGMOID, c.g_fush, 2 * CH, c.wg_h + (size_t)g * CH * 2 * CH,
                c.wg_b + (size_t)g * CH, c.g_w, CH, M, CH, 2 * CH, c.stream);
    { int total = M * 2 * CH;
      mix_kernel<<<(total + 255) / 256, 256, 0, c.stream>>>(t1, c.g_t2u, c.g_w, c.g_mixh, total); }
    launch_gemm(ACT_NONE, c.g_mixh, 2 * CH, c.db_h + (size_t)g * CH * 2 * CH,
                c.db_b + (size_t)g * CH, c.g_f, CH, M, CH, 2 * CH, c.stream);
    run_mb(c, c.g_f, out, mb_i, Lo);
}

extern "C" void kernel_launch(void* const* d_in, const int* in_sizes, int n_in,
                              void* d_out, int out_size, void* d_ws, size_t ws_size,
                              hipStream_t stream)
{
    (void)in_sizes; (void)n_in; (void)out_size; (void)ws_size;

    Ctx c;
    const float* x      = (const float*)d_in[0];
    const float* m_Win  = (const float*)d_in[1];
    c.m_convw = (const float*)d_in[2];
    c.m_convb = (const float*)d_in[3];
    const float* m_Wx   = (const float*)d_in[4];
    const float* m_Wdt  = (const float*)d_in[5];
    c.m_bdt   = (const float*)d_in[6];
    c.m_Alog  = (const float*)d_in[7];
    c.m_D     = (const float*)d_in[8];
    const float* m_Wout = (const float*)d_in[9];
    c.dc_w    = (const float*)d_in[10];
    c.dc_b    = (const float*)d_in[11];
    const float* wg_W   = (const float*)d_in[12];
    c.wg_b    = (const float*)d_in[13];
    const float* db_W   = (const float*)d_in[14];
    c.db_b    = (const float*)d_in[15];
    c.up_w    = (const float*)d_in[16];
    c.up_b    = (const float*)d_in[17];
    c.stream  = stream;

    // Bump allocator over the workspace (bytes, 256B-aligned chunks).
    char* base = (char*)d_ws;
    size_t off = 0;
    auto alloc = [&](size_t bytes) { void* r = base + off; off += (bytes + 255) & ~(size_t)255; return r; };

    float* e1 = (float*)alloc((size_t)BATCH * 1024 * CH * 4);
    float* e2 = (float*)alloc((size_t)BATCH * 512 * CH * 4);
    float* e3 = (float*)alloc((size_t)BATCH * 256 * CH * 4);
    float* e4 = (float*)alloc((size_t)BATCH * 128 * CH * 4);
    float* x2 = (float*)alloc((size_t)BATCH * 512 * CH * 4);
    float* x3 = (float*)alloc((size_t)BATCH * 256 * CH * 4);
    float* x4 = (float*)alloc((size_t)BATCH * 128 * CH * 4);
    float* d4 = (float*)alloc((size_t)BATCH * 256 * CH * 4);
    float* d3 = (float*)alloc((size_t)BATCH * 512 * CH * 4);

    _Float16* win_h  = (_Float16*)alloc((size_t)7 * 2 * DI * CH * 2);
    _Float16* wxp_h  = (_Float16*)alloc((size_t)7 * XDB_LD * DI * 2);
    _Float16* wdtp_h = (_Float16*)alloc((size_t)7 * DI * 32 * 2);
    _Float16* wout_h = (_Float16*)alloc((size_t)7 * CH * DI * 2);
    _Float16* wg_h   = (_Float16*)alloc((size_t)3 * CH * 2 * CH * 2);
    _Float16* db_h   = (_Float16*)alloc((size_t)3 * CH * 2 * CH * 2);
    c.win_h = win_h; c.wxp_h = wxp_h; c.wdtp_h = wdtp_h;
    c.wout_h = wout_h; c.wg_h = wg_h; c.db_h = db_h;

    c.t_xz    = (float*)alloc((size_t)BATCH * 1024 * 2 * DI * 4);
    c.t_u     = (float*)alloc((size_t)BATCH * 1024 * DI * 4);
    c.t_xdb   = (float*)alloc((size_t)BATCH * 1024 * XDB_LD * 4);
    c.t_dt    = (float*)alloc((size_t)BATCH * 1024 * DI * 4);
    c.t_y     = (float*)alloc((size_t)BATCH * 1024 * DI * 4);
    c.t_inh   = (_Float16*)alloc((size_t)BATCH * 1024 * CH * 2);
    c.t_uh    = (_Float16*)alloc((size_t)BATCH * 1024 * DI * 2);
    c.t_dtinh = (_Float16*)alloc((size_t)BATCH * 1024 * 32 * 2);
    c.t_yzh   = (_Float16*)alloc((size_t)BATCH * 1024 * DI * 2);
    c.g_fush  = (_Float16*)alloc((size_t)BATCH * 1024 * 2 * CH * 2);
    c.g_mixh  = (_Float16*)alloc((size_t)BATCH * 1024 * 2 * CH * 2);
    c.g_t2u   = (float*)alloc((size_t)BATCH * 1024 * CH * 4);
    c.g_w     = (float*)alloc((size_t)BATCH * 1024 * CH * 4);
    c.g_f     = (float*)alloc((size_t)BATCH * 1024 * CH * 4);

    // One-time weight conversion / padding to f16.
    { int t4 = (7 * 2 * DI * CH) / 4;
      cvt_f32_f16_kernel<<<(t4 + 255) / 256, 256, 0, stream>>>(m_Win, win_h, t4); }
    { int t4 = (7 * CH * DI) / 4;
      cvt_f32_f16_kernel<<<(t4 + 255) / 256, 256, 0, stream>>>(m_Wout, wout_h, t4); }
    { int t4 = (3 * CH * 2 * CH) / 4;
      cvt_f32_f16_kernel<<<(t4 + 255) / 256, 256, 0, stream>>>(wg_W, wg_h, t4);
      cvt_f32_f16_kernel<<<(t4 + 255) / 256, 256, 0, stream>>>(db_W, db_h, t4); }
    { int total = 7 * XDB_LD * DI;
      pad_wx_kernel<<<(total + 255) / 256, 256, 0, stream>>>(m_Wx, wxp_h, total); }
    { int total = 7 * DI * 32;
      pad_wdt_kernel<<<(total + 255) / 256, 256, 0, stream>>>(m_Wdt, wdtp_h, total); }

    // Encoder (all activations kept token-major [B, L, C])
    run_mb(c, x, e1, 0, 1024);
    { int Lo = 512, total = BATCH * Lo * CH;
      downconv_kernel<<<(total + 255) / 256, 256, 0, stream>>>(
          x, c.dc_w + 0 * (size_t)CH * CH * 3, c.dc_b + 0 * CH, x2, 1024, Lo, total); }
    run_mb(c, x2, e2, 1, 512);
    { int Lo = 256, total = BATCH * Lo * CH;
      downconv_kernel<<<(total + 255) / 256, 256, 0, stream>>>(
          x2, c.dc_w + 1 * (size_t)CH * CH * 3, c.dc_b + 1 * CH, x3, 512, Lo, total); }
    run_mb(c, x3, e3, 2, 256);
    { int Lo = 128, total = BATCH * Lo * CH;
      downconv_kernel<<<(total + 255) / 256, 256, 0, stream>>>(
          x3, c.dc_w + 2 * (size_t)CH * CH * 3, c.dc_b + 2 * CH, x4, 256, Lo, total); }
    run_mb(c, x4, e4, 3, 128);

    // Decoder
    run_gate(c, e3, 256, e4, 0, 4, d4);
    run_gate(c, e2, 512, d4, 1, 5, d3);
    run_gate(c, e1, 1024, d3, 2, 6, (float*)d_out);
}
